// MultiHeadGraphAttention_18691697672327
// MI455X (gfx1250) — matrix operand order repaired and checked by reading, not compiled, not run on hardware
//
#include <hip/hip_runtime.h>
#include <hip/hip_bf16.h>

// ---------------------------------------------------------------------------
// MI455X (gfx1250) multi-head graph attention, bf16 WMMA + f32 accumulate.
// ~103 GFLOP vs ~200 MB min HBM traffic -> memory-roofline bound only on the
// v_wmma_f32_16x16x32_bf16 path. V is produced in (B,H,DH,T) transposed layout
// so the PV B-fragments are contiguous global b128 loads (no LDS transpose,
// no barriers in the attention inner loop). The output-projection GEMM stages
// its bf16 X tile through the Tensor Data Mover (tensor_load_to_lds +
// s_wait_tensorcnt) when the builtin is available.
// ---------------------------------------------------------------------------

#define DEV __device__ __forceinline__

typedef __attribute__((ext_vector_type(16))) __bf16        bf16x16;
typedef __attribute__((ext_vector_type(2)))  __bf16        bf16x2;
typedef __attribute__((ext_vector_type(8)))  float         f32x8;
typedef __attribute__((ext_vector_type(4)))  float         f32x4;
typedef __attribute__((ext_vector_type(4)))  unsigned int  u32x4;
typedef __attribute__((ext_vector_type(2)))  unsigned int  u32x2;
typedef __attribute__((ext_vector_type(8)))  unsigned int  u32x8;
typedef __attribute__((ext_vector_type(4)))  int           i32x4;
typedef __attribute__((ext_vector_type(8)))  int           i32x8;

static constexpr int B_  = 8;
static constexpr int T_  = 1024;
static constexpr int D_  = 1024;
static constexpr int H_  = 16;
static constexpr int DH_ = 64;

#if __has_builtin(__builtin_amdgcn_tensor_load_to_lds) && \
    __has_builtin(__builtin_amdgcn_s_wait_tensorcnt)
#define TDM_OK 1
#endif

DEV unsigned short f32_to_bf16(float f) {
  unsigned int u = __builtin_bit_cast(unsigned int, f);
  u += 0x7FFFu + ((u >> 16) & 1u);            // round to nearest even
  return (unsigned short)(u >> 16);
}

// Pack two f32 -> packed bf16x2 dword (v_cvt_pk_bf16_f32 when available).
DEV unsigned int pack_bf16(float a, float b) {
#if __has_builtin(__builtin_amdgcn_cvt_pk_bf16_f32)
  bf16x2 p = __builtin_amdgcn_cvt_pk_bf16_f32(a, b);
  return __builtin_bit_cast(unsigned int, p);
#else
  unsigned int ua = __builtin_bit_cast(unsigned int, a);
  unsigned int ub = __builtin_bit_cast(unsigned int, b);
  ua += 0x7FFFu + ((ua >> 16) & 1u);
  ub += 0x7FFFu + ((ub >> 16) & 1u);
  return (ua >> 16) | (ub & 0xFFFF0000u);
#endif
}

// Build a 16x bf16 fragment from two 16-byte aligned chunks.
// A-frag (16x32): chunk0 = k0+16h .. +7, chunk1 = k0+16h+8 .. +7   (h = lane>>4)
// B-frag (32x16): chunk0 = 16h+0..7,    chunk1 = 16h+8..15
DEV bf16x16 frag_from_2x16B(const unsigned short* p0, const unsigned short* p1) {
  u32x4 a = *(const u32x4*)p0;
  u32x4 b = *(const u32x4*)p1;
  u32x8 u;
  u[0] = a[0]; u[1] = a[1]; u[2] = a[2]; u[3] = a[3];
  u[4] = b[0]; u[5] = b[1]; u[6] = b[2]; u[7] = b[3];
  return __builtin_bit_cast(bf16x16, u);
}

DEV f32x8 wmma_bf16(bf16x16 a, bf16x16 b, f32x8 c) {
  // 8 args: (neg_a, A, neg_b, B, c_mod, C, reuse_a, reuse_b)
  return __builtin_amdgcn_wmma_f32_16x16x32_bf16(false, a, false, b,
                                                 (short)0, c, false, false);
}

#ifdef TDM_OK
// TDM 2D tile load: tile_d0 x tile_d1 elements of data_size=2B, tensor row
// stride `stride_el` elements, LDS rows padded (pad codes per D# group1:
// pad_interval 3 -> 16 dwords (= 32 shorts), pad_amount 7 -> 8 dwords
// (= 16 shorts) -> 48-short LDS row stride).
DEV void tdm_load_tile_bf16(const void* gptr, unsigned int lds_addr,
                            int tile_d0, int tile_d1, int tensor_d0,
                            int tensor_d1, int stride_el, int pad_int,
                            int pad_amt) {
  unsigned long long ga = (unsigned long long)(size_t)gptr;
  u32x4 g0;
  g0[0] = 1u;                                     // count=1, user descriptor
  g0[1] = lds_addr;                               // lds_addr [63:32]
  g0[2] = (unsigned int)ga;                       // global_addr low
  g0[3] = (unsigned int)(ga >> 32) | (2u << 30);  // global_addr hi | type=2
  i32x8 g1;
  g1[0] = (1 << 16)                    // data_size = 1 (2 bytes)
          | (1 << 20)                  // pad_enable
          | (pad_int << 22)            // pad_interval
          | (pad_amt << 25);           // pad_amount
  g1[1] = (int)((tensor_d0 & 0xFFFF) << 16);          // tensor_dim0 [63:48]
  g1[2] = (int)(((unsigned)tensor_d0 >> 16) |
                ((tensor_d1 & 0xFFFF) << 16));        // dim0 hi | dim1 lo
  g1[3] = (int)(((unsigned)tensor_d1 >> 16) |
                ((unsigned)tile_d0 << 16));           // dim1 hi | tile_dim0
  g1[4] = tile_d1;                                    // tile_dim1 (tile_dim2=0)
  g1[5] = stride_el;                                  // tensor_dim0_stride lo
  g1[6] = 0;
  g1[7] = 0;
  i32x4 z4 = {0, 0, 0, 0};
#if __clang_major__ >= 23
  i32x8 z8 = {0, 0, 0, 0, 0, 0, 0, 0};
  __builtin_amdgcn_tensor_load_to_lds(g0, g1, z4, z4, z8, 0);
#else
  __builtin_amdgcn_tensor_load_to_lds(g0, g1, z4, z4, 0);
#endif
}
#endif

// ---------------------------------------------------------------------------
// Kernel 1: projection GEMM  Y = X @ W + b   (X f32 8192x1024, W f32 1024x1024)
// Output bf16 in (B,H,T,DH) layout, or (B,H,DH,T) when vtrans!=0 (for V).
// Tile 128(M) x 64(N), K step 32, 8 waves: wave = one 16-row M strip x 4 Cfrags.
// ---------------------------------------------------------------------------
__global__ __launch_bounds__(256) void gemm_proj_kernel(
    const float* __restrict__ X, const float* __restrict__ W,
    const float* __restrict__ bias, unsigned short* __restrict__ out,
    int vtrans) {
  __shared__ __align__(16) unsigned short ldsX[128][48];  // row-major [m][k]
  __shared__ __align__(16) unsigned short ldsW[64][48];   // transposed [n][k]

  const int t = threadIdx.x;
  const int wave = t >> 5, lane = t & 31;
  const int lh = lane >> 4, l15 = lane & 15;
  const int m_tile = blockIdx.x * 128;
  const int n_tile = blockIdx.y * 64;

  const f32x8 zero = {0.f, 0.f, 0.f, 0.f, 0.f, 0.f, 0.f, 0.f};
  f32x8 acc[4];
#pragma unroll
  for (int nf = 0; nf < 4; ++nf) acc[nf] = zero;

  for (int k0 = 0; k0 < D_; k0 += 32) {
    // Phase 1: issue all b128 loads (X tile 128x32, W tile 32x64).
    f32x4 xv[4], wv[2];
#pragma unroll
    for (int i = 0; i < 4; ++i) {
      int idx = i * 256 + t;               // 0..1023 float4s
      int r = idx >> 3, c4 = (idx & 7) * 4;
      xv[i] = *(const f32x4*)&X[(size_t)(m_tile + r) * D_ + k0 + c4];
    }
#pragma unroll
    for (int i = 0; i < 2; ++i) {
      int idx = i * 256 + t;               // 0..511 float4s
      int kk = idx >> 4, n4 = (idx & 15) * 4;
      wv[i] = *(const f32x4*)&W[(size_t)(k0 + kk) * D_ + n_tile + n4];
    }
    // Prefetch next K tile (global_prefetch_b8).
    if (k0 + 32 < D_) {
      if (t < 128)
        __builtin_prefetch(&X[(size_t)(m_tile + t) * D_ + (k0 + 32)], 0, 1);
      else if (t < 160)
        __builtin_prefetch(&W[(size_t)(k0 + 32 + (t - 128)) * D_ + n_tile], 0, 1);
    }
    // Phase 2: convert and store to LDS.
#pragma unroll
    for (int i = 0; i < 4; ++i) {
      int idx = i * 256 + t;
      int r = idx >> 3, c4 = (idx & 7) * 4;
      u32x2 p;
      p[0] = pack_bf16(xv[i][0], xv[i][1]);
      p[1] = pack_bf16(xv[i][2], xv[i][3]);
      *(u32x2*)&ldsX[r][c4] = p;
    }
#pragma unroll
    for (int i = 0; i < 2; ++i) {
      int idx = i * 256 + t;
      int kk = idx >> 4, n4 = (idx & 15) * 4;
      ldsW[n4 + 0][kk] = f32_to_bf16(wv[i][0]);
      ldsW[n4 + 1][kk] = f32_to_bf16(wv[i][1]);
      ldsW[n4 + 2][kk] = f32_to_bf16(wv[i][2]);
      ldsW[n4 + 3][kk] = f32_to_bf16(wv[i][3]);
    }
    __syncthreads();

    bf16x16 afrag = frag_from_2x16B(&ldsX[wave * 16 + l15][lh * 16],
                                    &ldsX[wave * 16 + l15][lh * 16 + 8]);
#pragma unroll
    for (int nf = 0; nf < 4; ++nf) {
      bf16x16 bfrag = frag_from_2x16B(&ldsW[nf * 16 + l15][lh * 16],
                                      &ldsW[nf * 16 + l15][lh * 16 + 8]);
      acc[nf] = wmma_bf16(afrag, bfrag, acc[nf]);
    }
    __syncthreads();
  }

  // Epilogue: +bias, write bf16. C layout: VGPR r -> M = r + 8h, N = lane&15.
  const int m_base = m_tile + wave * 16 + lh * 8;
#pragma unroll
  for (int nf = 0; nf < 4; ++nf) {
    int n = n_tile + nf * 16 + l15;
    float bb = bias[n];
    int head = n >> 6, dh = n & 63;
#pragma unroll
    for (int r = 0; r < 8; ++r) {
      int m = m_base + r;
      int bi = m >> 10, tt = m & 1023;
      size_t o = vtrans
                     ? ((size_t)(bi * H_ + head) * DH_ + dh) * T_ + tt
                     : ((size_t)(bi * H_ + head) * T_ + tt) * DH_ + dh;
      out[o] = f32_to_bf16(acc[nf][r] + bb);
    }
  }
}

// ---------------------------------------------------------------------------
// Kernel 2: flash attention, barrier-free. grid = (T/128, B*H), 8 waves, each
// wave owns 16 q rows and runs fully independently. Per 64-key tile:
//   S = Q K^T  (K row-major (key,dh) IS the 32x16 B layout -> global b128)
//   bias+mask, online softmax (16-lane shfl_xor reductions per C half)
//   O += P V   (V in (B,H,DH,T) -> B-frags are contiguous global b128;
//               P goes C->A layout through this wave's private LDS region)
// ---------------------------------------------------------------------------
__global__ __launch_bounds__(256) void attn_kernel(
    const unsigned short* __restrict__ Q, const unsigned short* __restrict__ K,
    const unsigned short* __restrict__ V, const float* __restrict__ sp,
    const float* __restrict__ ed, const unsigned char* __restrict__ mask,
    unsigned short* __restrict__ out) {
  __shared__ __align__(16) unsigned short ldsP[8][16][72];  // per-wave P tile

  const int t = threadIdx.x;
  const int wave = t >> 5, lane = t & 31;
  const int lh = lane >> 4, l15 = lane & 15;
  const int bh = blockIdx.y;          // 0..127
  const int b = bh >> 4, head = bh & 15;
  const size_t kbase = (size_t)bh * T_ * DH_;   // K: (bh, key, dh)
  const size_t vbase = (size_t)bh * DH_ * T_;   // V: (bh, dh, key)
  const int q0 = blockIdx.x * 128 + wave * 16;
  const float scale = 0.125f;         // 1/sqrt(64)

  // Q strip 16x64 -> two A fragments (K=0..31, 32..63) from global.
  bf16x16 qf[2];
#pragma unroll
  for (int ks = 0; ks < 2; ++ks) {
    const unsigned short* p = &Q[kbase + (size_t)(q0 + l15) * DH_ + ks * 32];
    qf[ks] = frag_from_2x16B(p + lh * 16, p + lh * 16 + 8);
  }

  const f32x8 zero = {0.f, 0.f, 0.f, 0.f, 0.f, 0.f, 0.f, 0.f};
  float rm[8], rs[8];
  f32x8 oacc[4];
#pragma unroll
  for (int r = 0; r < 8; ++r) { rm[r] = -3.0e38f; rs[r] = 0.f; }
#pragma unroll
  for (int nf = 0; nf < 4; ++nf) oacc[nf] = zero;

  for (int n0 = 0; n0 < T_; n0 += 64) {
    // Prefetch next kv tile.
    if (n0 + 64 < T_) {
      __builtin_prefetch(&K[kbase + (size_t)(n0 + 64 + (lane & 63)) * DH_], 0, 1);
      __builtin_prefetch(&V[vbase + (size_t)l15 * T_ + n0 + 64], 0, 1);
    }

    // Scores: 4 N-frags x 2 K-steps of WMMA; K B-frags direct from global.
    f32x8 s[4];
#pragma unroll
    for (int nf = 0; nf < 4; ++nf) {
      s[nf] = zero;
      const unsigned short* kp = &K[kbase + (size_t)(n0 + nf * 16 + l15) * DH_];
#pragma unroll
      for (int ks = 0; ks < 2; ++ks) {
        bf16x16 bf = frag_from_2x16B(kp + ks * 32 + lh * 16,
                                     kp + ks * 32 + lh * 16 + 8);
        s[nf] = wmma_bf16(qf[ks], bf, s[nf]);
      }
    }

    // scale + (spatial+edge) bias + key-padding mask; per-tile row max.
    float tmax[8];
#pragma unroll
    for (int r = 0; r < 8; ++r) tmax[r] = -3.0e38f;
#pragma unroll
    for (int nf = 0; nf < 4; ++nf) {
      int kcol = n0 + nf * 16 + l15;
      bool msk = mask[b * T_ + kcol] != 0;
      size_t rb = (size_t)b * T_ * T_ + kcol;
#pragma unroll
      for (int r = 0; r < 8; ++r) {
        int q = q0 + lh * 8 + r;
        float bv = msk ? -1.0e30f
                       : (sp[rb + (size_t)q * T_] + ed[rb + (size_t)q * T_]);
        float v = s[nf][r] * scale + bv;
        s[nf][r] = v;
        tmax[r] = fmaxf(tmax[r], v);
      }
    }
    // Row max across the 16 lanes of each half (rows differ per half).
#pragma unroll
    for (int r = 0; r < 8; ++r) {
      float v = tmax[r];
      v = fmaxf(v, __shfl_xor(v, 1, 32));
      v = fmaxf(v, __shfl_xor(v, 2, 32));
      v = fmaxf(v, __shfl_xor(v, 4, 32));
      v = fmaxf(v, __shfl_xor(v, 8, 32));
      tmax[r] = v;
    }
    // Online softmax rescale.
    float tsum[8];
#pragma unroll
    for (int r = 0; r < 8; ++r) {
      float mnew = fmaxf(rm[r], tmax[r]);
      float alpha = __expf(rm[r] - mnew);
      rm[r] = mnew;
      rs[r] *= alpha;
#pragma unroll
      for (int nf = 0; nf < 4; ++nf) oacc[nf][r] *= alpha;
      tsum[r] = 0.f;
    }
    // P = exp(s - m); write bf16 into this wave's LDS region (C -> A layout).
#pragma unroll
    for (int nf = 0; nf < 4; ++nf) {
#pragma unroll
      for (int r = 0; r < 8; ++r) {
        float p = __expf(s[nf][r] - rm[r]);
        tsum[r] += p;
        ldsP[wave][lh * 8 + r][nf * 16 + l15] = f32_to_bf16(p);
      }
    }
#pragma unroll
    for (int r = 0; r < 8; ++r) {
      float v = tsum[r];
      v += __shfl_xor(v, 1, 32);
      v += __shfl_xor(v, 2, 32);
      v += __shfl_xor(v, 4, 32);
      v += __shfl_xor(v, 8, 32);
      rs[r] += v;
    }
    // O += P x V. Same-wave LDS write->read: DS ops are in-order per wave.
#pragma unroll
    for (int ks = 0; ks < 2; ++ks) {
      bf16x16 pf = frag_from_2x16B(&ldsP[wave][l15][ks * 32 + lh * 16],
                                   &ldsP[wave][l15][ks * 32 + lh * 16 + 8]);
#pragma unroll
      for (int nf = 0; nf < 4; ++nf) {
        const unsigned short* vp =
            &V[vbase + (size_t)(nf * 16 + l15) * T_ + n0 + ks * 32 + lh * 16];
        bf16x16 vf = frag_from_2x16B(vp, vp + 8);
        oacc[nf] = wmma_bf16(pf, vf, oacc[nf]);
      }
    }
  }

  // Normalize and write (B, T, D) bf16.
#pragma unroll
  for (int r = 0; r < 8; ++r) {
    float inv = 1.0f / rs[r];
    int q = q0 + lh * 8 + r;
    size_t ob = ((size_t)b * T_ + q) * D_ + head * DH_;
#pragma unroll
    for (int nf = 0; nf < 4; ++nf)
      out[ob + nf * 16 + l15] = f32_to_bf16(oacc[nf][r] * inv);
  }
}

// ---------------------------------------------------------------------------
// Kernel 3: output projection. X bf16 (8192x1024) @ Wo (f32->bf16) + bo -> f32.
// X tile staged via the Tensor Data Mover when available (pure bf16 copy with
// padded LDS rows); wave 0 issues the descriptor, waits TENSORcnt, then the
// workgroup barrier publishes the tile.
// ---------------------------------------------------------------------------
__global__ __launch_bounds__(256) void gemm_out_kernel(
    const unsigned short* __restrict__ X, const float* __restrict__ W,
    const float* __restrict__ bias, float* __restrict__ out) {
  __shared__ __align__(16) unsigned short ldsX[128][48];
  __shared__ __align__(16) unsigned short ldsW[64][48];

  const int t = threadIdx.x;
  const int wave = t >> 5, lane = t & 31;
  const int lh = lane >> 4, l15 = lane & 15;
  const int m_tile = blockIdx.x * 128;
  const int n_tile = blockIdx.y * 64;

  const f32x8 zero = {0.f, 0.f, 0.f, 0.f, 0.f, 0.f, 0.f, 0.f};
  f32x8 acc[4];
#pragma unroll
  for (int nf = 0; nf < 4; ++nf) acc[nf] = zero;

  for (int k0 = 0; k0 < D_; k0 += 32) {
    // Stage X tile (128x32 bf16).
#ifdef TDM_OK
    if (wave == 0) {
      // 32 elem/row tile, 128 rows, row stride 1024 elem; LDS rows padded
      // 32 shorts data + 16 shorts pad = 48-short stride.
      tdm_load_tile_bf16(&X[(size_t)m_tile * D_ + k0],
                         (unsigned int)(size_t)&ldsX[0][0],
                         /*tile_d0=*/32, /*tile_d1=*/128,
                         /*tensor_d0=*/D_, /*tensor_d1=*/B_ * T_,
                         /*stride_el=*/D_, /*pad_int=*/3, /*pad_amt=*/7);
      __builtin_amdgcn_s_wait_tensorcnt(0);
    }
#else
#pragma unroll
    for (int i = 0; i < 2; ++i) {
      int idx = i * 256 + t;            // 0..511 8-short groups
      int r = idx >> 2, c8 = (idx & 3) * 8;
      *(u32x4*)&ldsX[r][c8] =
          *(const u32x4*)&X[(size_t)(m_tile + r) * D_ + k0 + c8];
    }
#endif
    // Stage W tile (32x64) transposed: split load / convert+store phases.
    f32x4 wv[2];
#pragma unroll
    for (int i = 0; i < 2; ++i) {
      int idx = i * 256 + t;            // 0..511 float4s
      int kk = idx >> 4, n4 = (idx & 15) * 4;
      wv[i] = *(const f32x4*)&W[(size_t)(k0 + kk) * D_ + n_tile + n4];
    }
    if (k0 + 32 < D_) {
      if (t < 128)
        __builtin_prefetch(&X[(size_t)(m_tile + t) * D_ + (k0 + 32)], 0, 1);
      else if (t < 160)
        __builtin_prefetch(&W[(size_t)(k0 + 32 + (t - 128)) * D_ + n_tile], 0, 1);
    }
#pragma unroll
    for (int i = 0; i < 2; ++i) {
      int idx = i * 256 + t;
      int kk = idx >> 4, n4 = (idx & 15) * 4;
      ldsW[n4 + 0][kk] = f32_to_bf16(wv[i][0]);
      ldsW[n4 + 1][kk] = f32_to_bf16(wv[i][1]);
      ldsW[n4 + 2][kk] = f32_to_bf16(wv[i][2]);
      ldsW[n4 + 3][kk] = f32_to_bf16(wv[i][3]);
    }
    __syncthreads();

    bf16x16 afrag = frag_from_2x16B(&ldsX[wave * 16 + l15][lh * 16],
                                    &ldsX[wave * 16 + l15][lh * 16 + 8]);
#pragma unroll
    for (int nf = 0; nf < 4; ++nf) {
      bf16x16 bfrag = frag_from_2x16B(&ldsW[nf * 16 + l15][lh * 16],
                                      &ldsW[nf * 16 + l15][lh * 16 + 8]);
      acc[nf] = wmma_bf16(afrag, bfrag, acc[nf]);
    }
    __syncthreads();
  }

  const int m_base = m_tile + wave * 16 + lh * 8;
#pragma unroll
  for (int nf = 0; nf < 4; ++nf) {
    int n = n_tile + nf * 16 + l15;
    float bb = bias[n];
#pragma unroll
    for (int r = 0; r < 8; ++r)
      out[(size_t)(m_base + r) * D_ + n] = acc[nf][r] + bb;
  }
}

// ---------------------------------------------------------------------------
extern "C" void kernel_launch(void* const* d_in, const int* in_sizes, int n_in,
                              void* d_out, int out_size, void* d_ws,
                              size_t ws_size, hipStream_t stream) {
  const float* query = (const float*)d_in[0];
  const float* key   = (const float*)d_in[1];
  const float* value = (const float*)d_in[2];
  const float* sp    = (const float*)d_in[3];
  const float* ed    = (const float*)d_in[4];
  const unsigned char* mask = (const unsigned char*)d_in[5];
  const float* Wq = (const float*)d_in[6];
  const float* bq = (const float*)d_in[7];
  const float* Wk = (const float*)d_in[8];
  const float* bk = (const float*)d_in[9];
  const float* Wv = (const float*)d_in[10];
  const float* bv = (const float*)d_in[11];
  const float* Wo = (const float*)d_in[12];
  const float* bo = (const float*)d_in[13];

  // Workspace: Q,K bf16 (B,H,T,DH); V bf16 (B,H,DH,T); attn out bf16 (B,T,D).
  const size_t QKV_ELEMS = (size_t)B_ * H_ * T_ * DH_;  // 8,388,608
  unsigned short* qw = (unsigned short*)d_ws;
  unsigned short* kw = qw + QKV_ELEMS;
  unsigned short* vw = kw + QKV_ELEMS;
  unsigned short* aw = vw + QKV_ELEMS;   // total 64 MB

  dim3 gg(64, 16), gb(256);
  gemm_proj_kernel<<<gg, gb, 0, stream>>>(query, Wq, bq, qw, 0);
  gemm_proj_kernel<<<gg, gb, 0, stream>>>(key,   Wk, bk, kw, 0);
  gemm_proj_kernel<<<gg, gb, 0, stream>>>(value, Wv, bv, vw, 1);

  attn_kernel<<<dim3(T_ / 128, B_ * H_), 256, 0, stream>>>(qw, kw, vw, sp, ed,
                                                           mask, aw);

  gemm_out_kernel<<<gg, gb, 0, stream>>>(aw, Wo, bo, (float*)d_out);
}
